// WordLevelAttention_3289944949530
// MI455X (gfx1250) — compile-verified, matrix operand order checked by reading, not run on hardware
//
#include <hip/hip_runtime.h>
#include <hip/hip_bf16.h>
#include <math.h>

// Problem dimensions (fixed by the reference)
constexpr int BB = 64;
constexpr int SS = 2048;
constexpr int EE = 512;
constexpr int CC = 512;
constexpr int HH = 256;

typedef __attribute__((ext_vector_type(16))) __bf16   v16bf;
typedef __attribute__((ext_vector_type(8)))  float    v8f;
typedef __attribute__((ext_vector_type(8)))  unsigned v8u;

#if __has_builtin(__builtin_amdgcn_tanhf)
#define TANHF(x) __builtin_amdgcn_tanhf(x)   // native v_tanh_f32 on gfx1250
#else
#define TANHF(x) tanhf(x)
#endif

// f32 -> bf16 round-to-nearest-even (bit ops; no scalar __bf16 arithmetic)
__device__ __forceinline__ unsigned short f2bf(float f) {
    unsigned u = __builtin_bit_cast(unsigned, f);
    unsigned r = u + 0x7fffu + ((u >> 16) & 1u);
    return (unsigned short)(r >> 16);
}

// Load one 16-bit WMMA fragment half-K pattern: 4 dwords at +0 and 4 at +16
// (ISA 7.12.2 16-bit A/B layout: lanes<16 get K sub-block 0, lanes>=16 get +8)
__device__ __forceinline__ v8u load_frag8(const unsigned short* p) {
    v8u u;
    #pragma unroll
    for (int i = 0; i < 4; ++i) u[i]     = *(const unsigned*)(p + 2 * i);
    #pragma unroll
    for (int i = 0; i < 4; ++i) u[4 + i] = *(const unsigned*)(p + 16 + 2 * i);
    return u;
}

__device__ __forceinline__ v8f wmma_bf16(v8u a, v8u b, v8f c) {
    return __builtin_amdgcn_wmma_f32_16x16x32_bf16(
        false, __builtin_bit_cast(v16bf, a),
        false, __builtin_bit_cast(v16bf, b),
        (short)0, c, false, false);
}

// ---------------------------------------------------------------------------
// Kernel 0: convert W_word [H,E] f32 -> bf16 (workspace, 256 KB, L2-resident)
// ---------------------------------------------------------------------------
__global__ __launch_bounds__(256) void k_convW(const float* __restrict__ W,
                                               unsigned short* __restrict__ wbf) {
    int idx = blockIdx.x * 256 + threadIdx.x;   // grid covers H*E = 131072
    wbf[idx] = f2bf(W[idx]);
}

// ---------------------------------------------------------------------------
// Kernel 1: ctx_feat[b,h] = context[b,:] @ W_ctx[h,:] + b_ctx[h]
// ---------------------------------------------------------------------------
__global__ __launch_bounds__(256) void k_ctx(const float* __restrict__ ctx,
                                             const float* __restrict__ Wctx,
                                             const float* __restrict__ bctx,
                                             float* __restrict__ ctxf) {
    __shared__ float cv[CC];
    int b = blockIdx.x, tid = threadIdx.x;
    cv[tid]       = ctx[b * CC + tid];
    cv[tid + 256] = ctx[b * CC + tid + 256];
    __syncthreads();
    int h = tid;
    const float* wr = Wctx + (size_t)h * CC;
    float acc = bctx[h];
    #pragma unroll 4
    for (int c = 0; c < CC; ++c) acc = fmaf(cv[c], wr[c], acc);
    ctxf[b * HH + h] = acc;
}

// ---------------------------------------------------------------------------
// Kernel 2 (WMMA): scores[b, s0..s0+31].
// Block = 256 threads (8 waves), grid (S/32, B).
// A tile: 32 x 512 embeds staged f32->bf16 in LDS (33 KB).
// Wave w handles N-tiles {w, w+8}; per N-tile, 2 M-tiles reuse one B fragment.
// Two-stage ping-pong pipeline on B fragments (loads write their consumer
// variable directly -> no register rotation); both A fragments issued before
// each WMMA pair so the second's LDS latency hides under the first WMMA.
// ---------------------------------------------------------------------------
#define ASTRIDE 520   // bf16 elements per LDS row (512 + 8 pad)
#define MROWS   32

__global__ __launch_bounds__(256) void k_scores(const float* __restrict__ embeds,
                                                const unsigned short* __restrict__ wbf,
                                                const float* __restrict__ ctxf,
                                                const float* __restrict__ bword,
                                                const float* __restrict__ wh,
                                                float* __restrict__ scores) {
    __shared__ unsigned short At[MROWS * ASTRIDE];
    __shared__ float sred[MROWS];

    const int b   = blockIdx.y;
    const int s0  = blockIdx.x * MROWS;
    const int tid = threadIdx.x;

    if (tid < MROWS) sred[tid] = 0.0f;

    // ---- stage A tile: embeds[b, s0:s0+32, 0:512] f32 -> bf16 in LDS ----
    const float* src = embeds + ((size_t)b * SS + s0) * EE;
    #pragma unroll
    for (int t = 0; t < 16; ++t) {
        int li  = tid + t * 256;          // 0..4095 float4 chunks
        int row = li >> 7;                // / 128
        int k4  = (li & 127) << 2;
        const float4 v = *(const float4*)(src + row * EE + k4);
        unsigned lo = ((unsigned)f2bf(v.y) << 16) | f2bf(v.x);
        unsigned hi = ((unsigned)f2bf(v.w) << 16) | f2bf(v.z);
        *(uint2*)&At[row * ASTRIDE + k4] = make_uint2(lo, hi);
    }
    __syncthreads();

    const int wave = tid >> 5;
    const int lane = tid & 31;
    const int m    = lane & 15;               // A row within M-tile / B,C,D column
    const int kb   = (lane < 16) ? 0 : 8;     // K sub-block per ISA 16-bit layout

    const unsigned short* arow0 = &At[m * ASTRIDE + kb];          // M-tile 0
    const unsigned short* arow1 = &At[(16 + m) * ASTRIDE + kb];   // M-tile 1

    #pragma unroll
    for (int j = 0; j < 2; ++j) {
        const int hbase = (wave + 8 * j) * 16;
        const int h     = hbase + m;          // this lane's N column
        const float cadd = ctxf[b * HH + h] + bword[h];
        const float whv  = wh[h];

        v8f acc0 = (v8f){};
        v8f acc1 = (v8f){};
        const unsigned short* brow = wbf + (size_t)h * EE + kb;

        // two-stage software pipeline over K (step 64 = 2 WMMA K-chunks)
        v8u bu0 = load_frag8(brow);
        #pragma unroll 2
        for (int kk = 0; kk < EE; kk += 64) {
            v8u bu1 = load_frag8(brow + kk + 32);
            {
                v8u a0 = load_frag8(arow0 + kk);
                v8u a1 = load_frag8(arow1 + kk);
                acc0 = wmma_bf16(a0, bu0, acc0);
                acc1 = wmma_bf16(a1, bu0, acc1);
            }
            if (kk + 64 < EE) bu0 = load_frag8(brow + kk + 64);
            {
                v8u a0 = load_frag8(arow0 + kk + 32);
                v8u a1 = load_frag8(arow1 + kk + 32);
                acc0 = wmma_bf16(a0, bu1, acc0);
                acc1 = wmma_bf16(a1, bu1, acc1);
            }
        }

        // acc{mt}[r] = word_feat[M, N], M = mt*16 + r (+8 for lanes 16..31),
        // N = h.  score[M] += tanh(word_feat + ctx + b_word) * w_hidden[N]
        #pragma unroll
        for (int mt = 0; mt < 2; ++mt) {
            #pragma unroll
            for (int r = 0; r < 8; ++r) {
                float v = TANHF((mt ? acc1[r] : acc0[r]) + cadd) * whv;
                v += __shfl_xor(v, 1, 32);
                v += __shfl_xor(v, 2, 32);
                v += __shfl_xor(v, 4, 32);
                v += __shfl_xor(v, 8, 32);
                if (m == 0) {
                    int mm = mt * 16 + r + ((lane < 16) ? 0 : 8);
                    atomicAdd(&sred[mm], v);     // ds_add_f32
                }
            }
        }
    }
    __syncthreads();
    if (tid < MROWS) scores[(size_t)b * SS + s0 + tid] = sred[tid];
}

// ---------------------------------------------------------------------------
// Kernel 3: masked softmax + renormalize. One block per batch, 256 thr x 8 el.
// ---------------------------------------------------------------------------
__global__ __launch_bounds__(256) void k_softmax(const float* __restrict__ scores,
                                                 const int* __restrict__ lens,
                                                 float* __restrict__ attn) {
    __shared__ float red[256];
    const int b = blockIdx.x, tid = threadIdx.x;
    const int len = lens[b];
    const float* sc = scores + (size_t)b * SS;

    float sv[8];
    float mx = -INFINITY;
    #pragma unroll
    for (int i = 0; i < 8; ++i) {
        sv[i] = sc[tid + i * 256];
        mx = fmaxf(mx, sv[i]);
    }
    red[tid] = mx; __syncthreads();
    for (int off = 128; off > 0; off >>= 1) {
        if (tid < off) red[tid] = fmaxf(red[tid], red[tid + off]);
        __syncthreads();
    }
    mx = red[0]; __syncthreads();

    float ev[8];
    float se = 0.0f, sm = 0.0f;
    #pragma unroll
    for (int i = 0; i < 8; ++i) {
        int s = tid + i * 256;
        float e = __expf(sv[i] - mx);      // v_exp_f32
        ev[i] = e;
        se += e;
        if (s < len) sm += e;
    }
    red[tid] = se; __syncthreads();
    for (int off = 128; off > 0; off >>= 1) {
        if (tid < off) red[tid] += red[tid + off];
        __syncthreads();
    }
    se = red[0]; __syncthreads();

    red[tid] = sm; __syncthreads();
    for (int off = 128; off > 0; off >>= 1) {
        if (tid < off) red[tid] += red[tid + off];
        __syncthreads();
    }
    sm = red[0];

    const float inv   = 1.0f / se;
    const float denom = 1e-10f + sm * inv;
    const float scale = inv / denom;
    #pragma unroll
    for (int i = 0; i < 8; ++i) {
        int s = tid + i * 256;
        attn[(size_t)b * SS + s] = (s < len) ? ev[i] * scale : 0.0f;
    }
}

// ---------------------------------------------------------------------------
// Kernel 4a: zero attention_features (d_out is poisoned by the harness)
// ---------------------------------------------------------------------------
__global__ __launch_bounds__(256) void k_zero(float* __restrict__ out) {
    out[blockIdx.x * 256 + threadIdx.x] = 0.0f;   // grid covers B*E = 32768
}

// ---------------------------------------------------------------------------
// Kernel 4b: attention_features[b,e] += sum_{s in chunk} attn[b,s]*embeds[b,s,e]
// grid (B, 16) chunks of 128 seq rows; 128 thr x float4 over E; f32 atomics.
// ---------------------------------------------------------------------------
__device__ __forceinline__ void fma4(float4& a, float s, const float4& v) {
    a.x = fmaf(s, v.x, a.x);
    a.y = fmaf(s, v.y, a.y);
    a.z = fmaf(s, v.z, a.z);
    a.w = fmaf(s, v.w, a.w);
}

__global__ __launch_bounds__(128) void k_wsum(const float* __restrict__ embeds,
                                              const float* __restrict__ attn,
                                              float* __restrict__ out) {
    const int b     = blockIdx.x;
    const int chunk = blockIdx.y;
    const int s0    = chunk * (SS / 16);          // 128 rows per chunk
    const int e4    = threadIdx.x * 4;
    const float* ab   = attn + (size_t)b * SS + s0;
    const float* base = embeds + ((size_t)b * SS + s0) * EE + e4;

    float4 acc = make_float4(0.f, 0.f, 0.f, 0.f);
    for (int s = 0; s < SS / 16; s += 4) {
        float4 av = *(const float4*)(ab + s);     // uniform -> scalar loads
        fma4(acc, av.x, *(const float4*)(base + (size_t)(s + 0) * EE));
        fma4(acc, av.y, *(const float4*)(base + (size_t)(s + 1) * EE));
        fma4(acc, av.z, *(const float4*)(base + (size_t)(s + 2) * EE));
        fma4(acc, av.w, *(const float4*)(base + (size_t)(s + 3) * EE));
    }
    float* o = out + b * EE + e4;
    atomicAdd(o + 0, acc.x);                      // global_atomic_add_f32
    atomicAdd(o + 1, acc.y);
    atomicAdd(o + 2, acc.z);
    atomicAdd(o + 3, acc.w);
}

// ---------------------------------------------------------------------------
extern "C" void kernel_launch(void* const* d_in, const int* in_sizes, int n_in,
                              void* d_out, int out_size, void* d_ws, size_t ws_size,
                              hipStream_t stream) {
    const float* embeds = (const float*)d_in[0];   // [B,S,E]
    const int*   lens   = (const int*)  d_in[1];   // [B]
    const float* ctx    = (const float*)d_in[2];   // [B,C]
    const float* Wword  = (const float*)d_in[3];   // [H,E]
    const float* bword  = (const float*)d_in[4];   // [H]
    const float* Wctx   = (const float*)d_in[5];   // [H,C]
    const float* bctx   = (const float*)d_in[6];   // [H]
    const float* wh     = (const float*)d_in[7];   // [H]

    // d_out = attention_features [B,E] f32, then attn [B,S] f32
    float* out_feat = (float*)d_out;
    float* out_attn = (float*)d_out + (size_t)BB * EE;

    // workspace partition
    unsigned short* wbf = (unsigned short*)d_ws;                        // 256 KB
    float* ctxf         = (float*)((char*)d_ws + (size_t)HH * EE * 2);  //  64 KB
    float* scores       = (float*)((char*)ctxf + (size_t)BB * HH * 4);  // 512 KB

    k_convW  <<<(HH * EE) / 256, 256, 0, stream>>>(Wword, wbf);
    k_ctx    <<<BB, 256, 0, stream>>>(ctx, Wctx, bctx, ctxf);
    k_zero   <<<(BB * EE) / 256, 256, 0, stream>>>(out_feat);
    k_scores <<<dim3(SS / MROWS, BB), 256, 0, stream>>>(embeds, wbf, ctxf, bword, wh, scores);
    k_softmax<<<BB, 256, 0, stream>>>(scores, lens, out_attn);
    k_wsum   <<<dim3(BB, 16), 128, 0, stream>>>(embeds, out_attn, out_feat);
}
